// VectorQuantizer_21053929685349
// MI455X (gfx1250) — matrix operand order repaired and checked by reading, not compiled, not run on hardware
//
#include <hip/hip_runtime.h>

typedef _Float16 v16h __attribute__((ext_vector_type(16)));
typedef float    v8f  __attribute__((ext_vector_type(8)));

#define N_E   1024
#define E_DIM 256
#define HWN   4096            // 64*64
#define NBAT  16
#define NTOT  (NBAT * HWN)    // 65536 vectors
#define CHW   (E_DIM * HWN)   // per-batch stride in z (floats)
#define BETA  0.25f
#define DECAY 0.99f
#define OMD   0.01f
#define EPSV  1e-5f

// ---------------- ws layout (bytes) ----------------
#define WS_ENORM  524288
#define WS_IDX    528384
#define WS_COUNTS 790528
#define WS_DW     794624
#define WS_LOSS   1843200
#define WS_NSUM   1843204

// ---------------- d_out layout (floats) ----------------
#define DO_LOSS 16777216
#define DO_IDX  16777217
#define DO_NCS  16842753
#define DO_NAVG 16843777
#define DO_NW   17105921

// ---- CDNA5 async global->LDS (param type learned from round-2 diagnostic:
//      non-const AS1 pointer to gcc-vector int4; LDS side AS3) ----
#if defined(__has_builtin)
# if __has_builtin(__builtin_amdgcn_global_load_async_to_lds_b128) && \
     __has_builtin(__builtin_amdgcn_s_wait_asynccnt)
#  define HAVE_ASYNC 1
# endif
#endif
#ifndef HAVE_ASYNC
# define HAVE_ASYNC 0
#endif

#if HAVE_ASYNC
typedef int v4i_t __attribute__((vector_size(4 * sizeof(int))));
typedef __attribute__((address_space(1))) v4i_t* gv4_p;
typedef __attribute__((address_space(3))) v4i_t* lv4_p;
#endif

// -------- zero the per-call accumulators --------
__global__ __launch_bounds__(256) void vq_zero(float* counts, float* dw, float* losss) {
    int i = blockIdx.x * 256 + threadIdx.x;
    if (i < N_E) counts[i] = 0.0f;
    if (i < N_E * E_DIM) dw[i] = 0.0f;
    if (i == 0) losss[0] = 0.0f;
}

// -------- pre-swizzle codebook into WMMA B-fragment layout + ||e||^2 --------
// Fragment f = tile*8 + kc covers codes [16*tile,16*tile+16) x K [32*kc,32*kc+32).
// B 32x16 (KxN), 16-bit: lanes 0-15 -> N=lane, K={0..7,16..23}; lanes 16-31 -> N=lane-16, K={8..15,24..31}.
__global__ __launch_bounds__(256) void vq_prep(const float* __restrict__ w,
                                               _Float16* __restrict__ wsB,
                                               float* __restrict__ enorm) {
    int gid = blockIdx.x * 256 + threadIdx.x;      // 0..16383 = frag*32 + lane
    int f = gid >> 5, lane = gid & 31;
    int tile = f >> 3, kc = f & 7;
    int col = tile * 16 + (lane & 15);
    int kb = kc * 32 + ((lane < 16) ? 0 : 8);
    _Float16* dst = wsB + (size_t)gid * 16;
#pragma unroll
    for (int i = 0; i < 16; ++i)
        dst[i] = (_Float16)w[(size_t)col * E_DIM + kb + (i < 8 ? i : i + 8)];

    if (gid < N_E) {
        float s = 0.0f;
        const float* wr = w + (size_t)gid * E_DIM;
        for (int c = 0; c < E_DIM; ++c) s += wr[c] * wr[c];
        enorm[gid] = s;
    }
}

// -------- main: fused f16-WMMA GEMM + argmin, LDS-shared B, double buffered --------
// Block = 8 waves = 8 row tiles (128 rows). One 8KB column tile of B is staged in
// LDS per step and shared by all 8 waves (8x less L2 traffic); next tile is
// fetched asynchronously while the current WMMA chain runs.
__global__ __launch_bounds__(256) void vq_assign(const float* __restrict__ z,
                                                 const _Float16* __restrict__ wsB,
                                                 const float* __restrict__ enorm,
                                                 int* __restrict__ wsidx,
                                                 float* __restrict__ counts,
                                                 float* __restrict__ idx_out) {
    __shared__ _Float16 sB[2][4096];    // 2 x 8192 B column tiles
    __shared__ float    s_en[N_E];      // ||e||^2

    int tid  = threadIdx.x;
    int lane = tid & 31;
    int wv   = tid >> 5;
    int n0   = (blockIdx.x * 8 + wv) * 16;          // this wave's 16-row tile
    int b    = n0 >> 12;
    int hw   = (n0 & 4095) + (lane & 15);
    const float* zb = z + (size_t)b * CHW + hw;
    int klo = (lane >> 4) * 8;  // A layout: lane<16 -> K {0..7,16..23}; lane>=16 -> {8..15,24..31}

    // stage enorm to LDS
    for (int i = tid; i < N_E; i += 256) s_en[i] = enorm[i];

    // stage column tile 0 to LDS (tile t = 256 consecutive v16h in wsB)
    const v16h* gB = (const v16h*)wsB;
    ((v16h*)sB[0])[tid] = gB[tid];

    // A fragments: 16 rows x K=256 resident in 64 VGPRs, converted f32->f16
    v16h a[8];
#pragma unroll
    for (int kc = 0; kc < 8; ++kc) {
#pragma unroll
        for (int i = 0; i < 16; ++i) {
            int k = kc * 32 + klo + (i < 8 ? i : i + 8);
            a[kc][i] = (_Float16)zb[(size_t)k * HWN];
        }
    }

    float minv[8];
    int   mini[8];
#pragma unroll
    for (int r = 0; r < 8; ++r) { minv[r] = 3.4e38f; mini[r] = 0; }

    int ncol = lane & 15;
    __syncthreads();

    for (int t = 0; t < 64; ++t) {
        int cur = t & 1;

        // kick off fetch of tile t+1 (uniform branch: EXEC stays all-ones)
#if HAVE_ASYNC
        if (t < 63) {
            const char* gsrc = (const char*)wsB + (size_t)(t + 1) * 8192 + (size_t)tid * 32;
            char*       ldst = (char*)&sB[cur ^ 1][0] + (size_t)tid * 32;
            gv4_p g1 = (gv4_p)(uintptr_t)gsrc;
            lv4_p l3 = (lv4_p)(uint32_t)(uintptr_t)ldst;
            __builtin_amdgcn_global_load_async_to_lds_b128(g1, l3, 0, 0);
            __builtin_amdgcn_global_load_async_to_lds_b128(g1, l3, 16, 0);
        }
#else
        v16h nxt;
        if (t < 63) nxt = gB[(t + 1) * 256 + tid];
#endif

        // preload this tile's 8 B fragments from LDS (clauses the ds_load_b128s)
        const v16h* lb = (const v16h*)sB[cur];
        v16h bf[8];
#pragma unroll
        for (int kc = 0; kc < 8; ++kc) bf[kc] = lb[kc * 32 + lane];

        v8f acc = {};
#pragma unroll
        for (int kc = 0; kc < 8; ++kc)
            acc = __builtin_amdgcn_wmma_f32_16x16x32_f16(false, a[kc], false, bf[kc],
                                                         (short)0, acc, false, false);

        float en = s_en[t * 16 + ncol];
#pragma unroll
        for (int r = 0; r < 8; ++r) {
            float sc = en - 2.0f * acc[r];   // ||z||^2 constant per row: dropped
            if (sc < minv[r]) { minv[r] = sc; mini[r] = t * 16 + ncol; }
        }

#if HAVE_ASYNC
        if (t < 63) __builtin_amdgcn_s_wait_asynccnt(0);
#else
        if (t < 63) ((v16h*)sB[cur ^ 1])[tid] = nxt;
#endif
        __syncthreads();
    }

    // reduce argmin across the 16 lanes holding each row (xor stays within half-wave)
#pragma unroll
    for (int m = 8; m >= 1; m >>= 1) {
#pragma unroll
        for (int r = 0; r < 8; ++r) {
            float ov = __shfl_xor(minv[r], m, 32);
            int   oi = __shfl_xor(mini[r], m, 32);
            if (ov < minv[r] || (ov == minv[r] && oi < mini[r])) { minv[r] = ov; mini[r] = oi; }
        }
    }
    if ((lane & 15) == 0) {
        int rbase = n0 + ((lane >> 4) ? 8 : 0);   // lane 0 -> rows 0..7, lane 16 -> rows 8..15
#pragma unroll
        for (int r = 0; r < 8; ++r) {
            wsidx[rbase + r]   = mini[r];
            idx_out[rbase + r] = (float)mini[r];
            atomicAdd(&counts[mini[r]], 1.0f);
        }
    }
}

// -------- gather z_q -> out (NCHW), loss partials, dw scatter --------
__global__ __launch_bounds__(256) void vq_scatter(const float* __restrict__ z,
                                                  const float* __restrict__ w,
                                                  const int* __restrict__ wsidx,
                                                  float* __restrict__ out,
                                                  float* __restrict__ dw,
                                                  float* __restrict__ losss) {
    __shared__ float red[256];
    int tid = threadIdx.x;
    int tx = tid & 63, ty = tid >> 6;
    int n = blockIdx.x * 64 + tx;                 // 64 consecutive hw: coalesced z/out
    int b = n >> 12, hw = n & 4095;
    int code = wsidx[n];
    const float* zb  = z + (size_t)b * CHW + hw;
    float*       ob  = out + (size_t)b * CHW + hw;
    const float* wr  = w  + (size_t)code * E_DIM;
    float*       dwr = dw + (size_t)code * E_DIM;
    float ls = 0.0f;
    for (int c = ty; c < E_DIM; c += 4) {
        float zv = zb[(size_t)c * HWN];
        float wv = wr[c];
        ob[(size_t)c * HWN] = wv;                 // straight-through output == z_q
        float d = wv - zv;
        ls += d * d;
        atomicAdd(&dwr[c], zv);
    }
    red[tid] = ls;
    __syncthreads();
    for (int s = 128; s >= 1; s >>= 1) {
        if (tid < s) red[tid] += red[tid + s];
        __syncthreads();
    }
    if (tid == 0) atomicAdd(losss, red[0]);
}

// -------- EMA cluster size + total count --------
__global__ __launch_bounds__(1024) void vq_fin_cs(const float* __restrict__ cs,
                                                  const float* __restrict__ counts,
                                                  float* __restrict__ ncs_out,
                                                  float* __restrict__ nsum) {
    __shared__ float red[1024];
    int j = threadIdx.x;
    float v = cs[j] * DECAY + OMD * counts[j];
    ncs_out[j] = v;
    red[j] = v;
    __syncthreads();
    for (int s = 512; s >= 1; s >>= 1) {
        if (j < s) red[j] += red[j + s];
        __syncthreads();
    }
    if (j == 0) nsum[0] = red[0];
}

// -------- EMA embedding avg, normalized weight, loss finalize --------
__global__ __launch_bounds__(256) void vq_fin_w(const float* __restrict__ avg,
                                                const float* __restrict__ dw,
                                                const float* __restrict__ ncs,
                                                const float* __restrict__ nsum,
                                                float* __restrict__ navg_out,
                                                float* __restrict__ nw_out,
                                                const float* __restrict__ losss,
                                                float* __restrict__ loss_out) {
    int i = blockIdx.x * 256 + threadIdx.x;       // 0..262143
    int j = i >> 8;
    float n = nsum[0];
    float css = (ncs[j] + EPSV) / (n + (float)N_E * EPSV) * n;
    float na = avg[i] * DECAY + OMD * dw[i];
    navg_out[i] = na;
    nw_out[i]   = na / css;
    if (i == 0) loss_out[0] = BETA * losss[0] / (float)(NBAT * E_DIM * HWN);
}

extern "C" void kernel_launch(void* const* d_in, const int* in_sizes, int n_in,
                              void* d_out, int out_size, void* d_ws, size_t ws_size,
                              hipStream_t stream) {
    const float* z      = (const float*)d_in[0];
    const float* weight = (const float*)d_in[1];
    const float* cs     = (const float*)d_in[2];
    const float* eavg   = (const float*)d_in[3];
    float* out = (float*)d_out;

    char* ws = (char*)d_ws;
    _Float16* wsB   = (_Float16*)ws;
    float* enorm    = (float*)(ws + WS_ENORM);
    int*   wsidx    = (int*)  (ws + WS_IDX);
    float* counts   = (float*)(ws + WS_COUNTS);
    float* dw       = (float*)(ws + WS_DW);
    float* losss    = (float*)(ws + WS_LOSS);
    float* nsum     = (float*)(ws + WS_NSUM);

    vq_zero<<<(N_E * E_DIM + 255) / 256, 256, 0, stream>>>(counts, dw, losss);
    vq_prep<<<64, 256, 0, stream>>>(weight, wsB, enorm);
    vq_assign<<<512, 256, 0, stream>>>(z, wsB, enorm, wsidx, counts, out + DO_IDX);
    vq_scatter<<<NTOT / 64, 256, 0, stream>>>(z, weight, wsidx, out, dw, losss);
    vq_fin_cs<<<1, 1024, 0, stream>>>(cs, counts, out + DO_NCS, nsum);
    vq_fin_w<<<(N_E * E_DIM + 255) / 256, 256, 0, stream>>>(
        eavg, dw, out + DO_NCS, nsum, out + DO_NAVG, out + DO_NW, losss, out + DO_LOSS);
}
